// SFMcell_32358283608670
// MI455X (gfx1250) — matrix operand order repaired and checked by reading, not compiled, not run on hardware
//
#include <hip/hip_runtime.h>
#include <hip/hip_bf16.h>
#include <math.h>

// ---------------------------------------------------------------------------
// Shapes (compile-time constants from the reference)
// ---------------------------------------------------------------------------
#define B_ROWS 2048
#define DIN    512
#define NS     256
#define NF     64
#define KX     768     // DIN + NS
#define NG     832     // 256 (fste) + 64 (ffreq) + 256 (i) + 256 (c)
#define KO     1024    // DIN + NS + NS

typedef __attribute__((ext_vector_type(8)))  __bf16 v8bf;
typedef __attribute__((ext_vector_type(16))) __bf16 v16bf;
typedef __attribute__((ext_vector_type(8)))  float  v8f;
typedef __attribute__((ext_vector_type(4)))  float  v4f;

__device__ __forceinline__ unsigned short f2bf(float f) {
  unsigned u = __float_as_uint(f);
  unsigned r = ((u >> 16) & 1u) + 0x7FFFu;   // round-to-nearest-even
  return (unsigned short)((u + r) >> 16);
}
__device__ __forceinline__ float sigmoidf_(float x) { return 1.0f / (1.0f + __expf(-x)); }

// ---------------------------------------------------------------------------
// CDNA5 WMMA fragment loaders (wave32, 16x16x32 bf16)
// A (16x32, MxK) per ISA 7.12.2: lanes 0-15 row M=lane hold K=[0..7],[16..23];
//                                lanes 16-31 row M=lane-16 hold K=[8..15],[24..31]
// B (32x16, KxN): lane holds column N=lane&15, same K order as A (K=[0..7],[16..23] / [8..15],[24..31])
// ---------------------------------------------------------------------------
__device__ __forceinline__ v16bf load_fragA(const __bf16* __restrict__ row, int kbase, int half) {
  const v8bf lo = *(const v8bf*)(row + kbase + half * 8);
  const v8bf hi = *(const v8bf*)(row + kbase + 16 + half * 8);
  v16bf r;
#pragma unroll
  for (int i = 0; i < 8; ++i) { r[i] = lo[i]; r[i + 8] = hi[i]; }
  return r;
}
__device__ __forceinline__ v16bf load_fragB(const __bf16* __restrict__ col, int kbase, int half) {
  const v8bf lo = *(const v8bf*)(col + kbase + half * 8);
  const v8bf hi = *(const v8bf*)(col + kbase + 16 + half * 8);
  v16bf r;
#pragma unroll
  for (int i = 0; i < 8; ++i) { r[i] = lo[i]; r[i + 8] = hi[i]; }
  return r;
}

// ---------------------------------------------------------------------------
// Prep kernels: bf16 packing of x_h and fused/transposed weights
// ---------------------------------------------------------------------------
__global__ void pack_xh(const float* __restrict__ x, const float* __restrict__ h,
                        unsigned short* __restrict__ xh) {
  int idx = blockIdx.x * blockDim.x + threadIdx.x;
  if (idx >= B_ROWS * KX) return;
  int b = idx / KX, k = idx - b * KX;
  float v = (k < DIN) ? x[b * DIN + k] : h[b * NS + (k - DIN)];
  xh[idx] = f2bf(v);
}

__global__ void pack_w(const float* __restrict__ Wfste, const float* __restrict__ Wffreq,
                       const float* __restrict__ Wi,    const float* __restrict__ Wc,
                       const float* __restrict__ Wo,
                       const float* __restrict__ bfste, const float* __restrict__ bffreq,
                       const float* __restrict__ bi,    const float* __restrict__ bc,
                       unsigned short* __restrict__ wt,   // [NG, KX] transposed fused gates
                       unsigned short* __restrict__ wot,  // [NS, KO] transposed W_o
                       float* __restrict__ ball) {        // [NG] fused bias
  const int NWT = NG * KX;
  const int NWO = NS * KO;
  int idx = blockIdx.x * blockDim.x + threadIdx.x;
  if (idx < NWT) {
    int n = idx / KX, k = idx - n * KX;
    float v;
    if      (n < 256) v = Wfste [k * 256 + n];
    else if (n < 320) v = Wffreq[k * 64  + (n - 256)];
    else if (n < 576) v = Wi    [k * 256 + (n - 320)];
    else              v = Wc    [k * 256 + (n - 576)];
    wt[idx] = f2bf(v);
  } else if (idx < NWT + NWO) {
    int j = idx - NWT;
    int n = j / KO, k = j - n * KO;
    wot[j] = f2bf(Wo[k * NS + n]);
  } else if (idx < NWT + NWO + NG) {
    int n = idx - NWT - NWO;
    float v;
    if      (n < 256) v = bfste [n];
    else if (n < 320) v = bffreq[n - 256];
    else if (n < 576) v = bi    [n - 320];
    else              v = bc    [n - 576];
    ball[n] = v;
  }
}

// ---------------------------------------------------------------------------
// Gates GEMM: G[2048, 832] = act(x_h @ W_all + b_all)
// block = 128 threads (4 waves); wave computes a 16x64 tile; grid (13, 32)
// Single-buffer fragment loop: the compiler pipelines/unrolls this cleanly
// (no spills); load latency is hidden by multi-wave occupancy.
// ---------------------------------------------------------------------------
__global__ void gemm_gates(const unsigned short* __restrict__ xh_u,
                           const unsigned short* __restrict__ wt_u,
                           const float* __restrict__ ball,
                           float* __restrict__ G) {
  const __bf16* xh = (const __bf16*)xh_u;
  const __bf16* wt = (const __bf16*)wt_u;
  int tid  = threadIdx.x;
  int wave = tid >> 5;
  int lane = tid & 31;
  int half = lane >> 4;
  int r    = lane & 15;
  int m0 = blockIdx.y * 64 + wave * 16;
  int n0 = blockIdx.x * 64;

  v8f acc[4] = {};
  const __bf16* arow = xh + (size_t)(m0 + r) * KX;
  for (int k0 = 0; k0 < KX; k0 += 32) {
    v16bf a = load_fragA(arow, k0, half);
#pragma unroll
    for (int nb = 0; nb < 4; ++nb) {
      v16bf b = load_fragB(wt + (size_t)(n0 + nb * 16 + r) * KX, k0, half);
      acc[nb] = __builtin_amdgcn_wmma_f32_16x16x32_bf16(
          false, a, false, b, (short)0, acc[nb], false, false);
    }
  }
#pragma unroll
  for (int nb = 0; nb < 4; ++nb) {
    int n = n0 + nb * 16 + r;
    float bias = ball[n];
#pragma unroll
    for (int v = 0; v < 8; ++v) {
      int m = m0 + v + 8 * half;
      float x = acc[nb][v] + bias;
      float y = (n < 576) ? sigmoidf_(x) : tanhf(x);
      G[(size_t)m * NG + n] = y;
    }
  }
}

// ---------------------------------------------------------------------------
// Fused S update (memory-bound pass, touches S once each way, non-temporal so
// the 536MB stream does not thrash the 192MB L2):
//   S' = S * f_ste*f_freq + (i*c)*omega ; A = sqrt(S0'^2+S1'^2)
//   c_t = sigmoid(A @ u_a + b_a)
// 16 lanes per (b,s) row, float4 per lane over NF=64
// ---------------------------------------------------------------------------
__global__ void supdate(const float* __restrict__ G, const float* __restrict__ Sin,
                        const float* __restrict__ ua, const float* __restrict__ ba,
                        const int* __restrict__ tptr,
                        float* __restrict__ Sout,            // d_out + 2048*256
                        float* __restrict__ ctf,             // [2048,256] f32
                        unsigned short* __restrict__ ctb) {  // [2048,256] bf16
  int gid   = blockIdx.x * blockDim.x + threadIdx.x;
  int group = gid >> 4;          // one (b, s) row
  int t16   = gid & 15;
  int b = group >> 8;
  int s = group & 255;
  int j0 = t16 * 4;
  int tt = *tptr;

  size_t gbase = (size_t)b * NG;
  float fste = G[gbase + s];
  v4f ff  = *(const v4f*)(G + gbase + 256 + j0);
  float ic   = G[gbase + 320 + s] * G[gbase + 576 + s];

  size_t so0 = (size_t)b * (2 * NS * NF) + (size_t)s * NF + j0;
  size_t so1 = so0 + (size_t)NS * NF;
  v4f s0 = __builtin_nontemporal_load((const v4f*)(Sin + so0));
  v4f s1 = __builtin_nontemporal_load((const v4f*)(Sin + so1));
  v4f ua4 = *(const v4f*)(ua + j0);

  v4f o0, o1;
  float acc = 0.0f;
#pragma unroll
  for (int i = 0; i < 4; ++i) {
    int j = j0 + i;
    float ang = (float)((j * tt) % NF) * 0.09817477042f;  // 2*pi/64
    float sn = __sinf(ang), cs = __cosf(ang);
    float f = fste * ff[i];
    float n0 = s0[i] * f + ic * sn;
    float n1 = s1[i] * f + ic * cs;
    o0[i] = n0; o1[i] = n1;
    acc += sqrtf(n0 * n0 + n1 * n1) * ua4[i];
  }
  __builtin_nontemporal_store(o0, (v4f*)(Sout + so0));
  __builtin_nontemporal_store(o1, (v4f*)(Sout + so1));

  // reduce over the 16 lanes covering this (b,s) row
#pragma unroll
  for (int off = 8; off > 0; off >>= 1) acc += __shfl_xor(acc, off, 16);
  if (t16 == 0) {
    float ct = sigmoidf_(acc + ba[s]);
    ctf[(size_t)b * NS + s] = ct;
    ctb[(size_t)b * NS + s] = f2bf(ct);
  }
}

// ---------------------------------------------------------------------------
// Output GEMM: h = sigmoid([x_h, c_t] @ W_o + b_o) * c_t   (K = 1024)
// A switches source at the k=768 boundary (uniform branch).
// ---------------------------------------------------------------------------
__global__ void gemm_out(const unsigned short* __restrict__ xh_u,
                         const unsigned short* __restrict__ ctb_u,
                         const unsigned short* __restrict__ wot_u,
                         const float* __restrict__ bo, const float* __restrict__ ctf,
                         float* __restrict__ hout) {
  const __bf16* xh  = (const __bf16*)xh_u;
  const __bf16* ctb = (const __bf16*)ctb_u;
  const __bf16* wot = (const __bf16*)wot_u;
  int tid  = threadIdx.x;
  int wave = tid >> 5;
  int lane = tid & 31;
  int half = lane >> 4;
  int r    = lane & 15;
  int m0 = blockIdx.y * 64 + wave * 16;
  int n0 = blockIdx.x * 64;

  const __bf16* axh = xh  + (size_t)(m0 + r) * KX;
  const __bf16* act = ctb + (size_t)(m0 + r) * NS;

  v8f acc[4] = {};
  for (int k0 = 0; k0 < KO; k0 += 32) {
    v16bf a = (k0 < KX) ? load_fragA(axh, k0, half)
                        : load_fragA(act, k0 - KX, half);
#pragma unroll
    for (int nb = 0; nb < 4; ++nb) {
      v16bf b = load_fragB(wot + (size_t)(n0 + nb * 16 + r) * KO, k0, half);
      acc[nb] = __builtin_amdgcn_wmma_f32_16x16x32_bf16(
          false, a, false, b, (short)0, acc[nb], false, false);
    }
  }
#pragma unroll
  for (int nb = 0; nb < 4; ++nb) {
    int n = n0 + nb * 16 + r;
    float bias = bo[n];
#pragma unroll
    for (int v = 0; v < 8; ++v) {
      int m = m0 + v + 8 * half;
      float o = sigmoidf_(acc[nb][v] + bias);
      hout[(size_t)m * NS + n] = o * ctf[(size_t)m * NS + n];
    }
  }
}

// ---------------------------------------------------------------------------
// Launch
// ---------------------------------------------------------------------------
extern "C" void kernel_launch(void* const* d_in, const int* in_sizes, int n_in,
                              void* d_out, int out_size, void* d_ws, size_t ws_size,
                              hipStream_t stream) {
  const float* x_i    = (const float*)d_in[0];
  const float* h_i    = (const float*)d_in[1];
  const float* S_i    = (const float*)d_in[2];
  const float* Wfste  = (const float*)d_in[3];
  const float* bfste  = (const float*)d_in[4];
  const float* Wffreq = (const float*)d_in[5];
  const float* bffreq = (const float*)d_in[6];
  const float* Wi     = (const float*)d_in[7];
  const float* bi     = (const float*)d_in[8];
  const float* Wc     = (const float*)d_in[9];
  const float* bc     = (const float*)d_in[10];
  const float* u_a    = (const float*)d_in[11];
  const float* b_a    = (const float*)d_in[12];
  const float* Wo     = (const float*)d_in[13];
  const float* bo     = (const float*)d_in[14];
  const int*   tptr   = (const int*)d_in[15];

  char* ws = (char*)d_ws;
  size_t off = 0;
  auto alloc = [&](size_t bytes) { size_t o = off; off += (bytes + 255) & ~(size_t)255; return o; };
  unsigned short* xh   = (unsigned short*)(ws + alloc((size_t)B_ROWS * KX * 2));
  unsigned short* wt   = (unsigned short*)(ws + alloc((size_t)NG * KX * 2));
  unsigned short* wot  = (unsigned short*)(ws + alloc((size_t)NS * KO * 2));
  float*          ball = (float*)         (ws + alloc((size_t)NG * 4));
  float*          G    = (float*)         (ws + alloc((size_t)B_ROWS * NG * 4));
  float*          ctf  = (float*)         (ws + alloc((size_t)B_ROWS * NS * 4));
  unsigned short* ctb  = (unsigned short*)(ws + alloc((size_t)B_ROWS * NS * 2));

  float* h_out = (float*)d_out;                         // [2048, 256]
  float* S_out = (float*)d_out + (size_t)B_ROWS * NS;   // [2048, 2, 256, 64]

  {
    int n = B_ROWS * KX;
    pack_xh<<<(n + 255) / 256, 256, 0, stream>>>(x_i, h_i, xh);
  }
  {
    int n = NG * KX + NS * KO + NG;
    pack_w<<<(n + 255) / 256, 256, 0, stream>>>(Wfste, Wffreq, Wi, Wc, Wo,
                                                bfste, bffreq, bi, bc, wt, wot, ball);
  }
  gemm_gates<<<dim3(NG / 64, B_ROWS / 64), 128, 0, stream>>>(xh, wt, ball, G);
  {
    int n = B_ROWS * NS * 16;  // 16 lanes per (b,s) row
    supdate<<<n / 256, 256, 0, stream>>>(G, S_i, u_a, b_a, tptr, S_out, ctf, ctb);
  }
  gemm_out<<<dim3(NS / 64, B_ROWS / 64), 128, 0, stream>>>(xh, ctb, wot, bo, ctf, h_out);
}
